// ConvSwitchedVAE_58720792871212
// MI455X (gfx1250) — matrix operand order repaired and checked by reading, not compiled, not run on hardware
//
#include <hip/hip_runtime.h>

// Sizes from the reference
#define BATCH 512

typedef _Float16 v8h  __attribute__((ext_vector_type(8)));
typedef _Float16 v16h __attribute__((ext_vector_type(16)));
typedef float    v8f  __attribute__((ext_vector_type(8)));
typedef unsigned int v4u __attribute__((ext_vector_type(4)));
typedef int v4i __attribute__((ext_vector_type(4)));
typedef int v8i __attribute__((ext_vector_type(8)));

// Output tuple layout (flat, in return order)
#define O_Y    16777216          // y_logits  (512*4)
#define O_IDX  (O_Y + 2048)      // idx       (512)
#define O_HARD (O_IDX + 512)     // y_hard    (512*4)
#define O_ZM   (O_HARD + 2048)   // z_mean_sel(512)
#define O_ZLV  (O_ZM + 512)      // z_logvar_sel
#define O_Z    (O_ZLV + 512)     // z_sel

// LDS layout for the big kernel (dynamic shared):
//   [0)          xraw : 32768 f32  (TDM destination, raw x[b])
//   [131072)     xpad : 34*34*32 f16 (relu(x), halo-padded, c contiguous)
//   ...          hpad : 34*34*8  f16
//   ...          w1t  : 9*16*32  f16
//   ...          w2t  : 32*96    f16
//   ...          b1s(8 f32), b2s(32 f32)
#define LDS_RAW_BYTES   131072
#define LDS_TOTAL_BYTES (LDS_RAW_BYTES +                                        \
                         (34 * 34 * 32 + 34 * 34 * 8 + 9 * 16 * 32 + 32 * 96) * \
                             2 +                                                \
                         8 * 4 + 32 * 4)

// ---------------------------------------------------------------------------
// Kernel 1: control path. One block per batch element.
// ---------------------------------------------------------------------------
__global__ __launch_bounds__(256) void vae_ctrl_kernel(
    const float* __restrict__ x, const float* __restrict__ gum,
    const float* __restrict__ eps, const float* __restrict__ Ws,
    const float* __restrict__ bs, const float* __restrict__ Wl,
    const float* __restrict__ bl, float* __restrict__ out,
    int* __restrict__ ws_idx, float* __restrict__ ws_scale) {
  __shared__ float sWs[512];
  __shared__ float sS[256];
  __shared__ float sCtrl[12];
  const int b = blockIdx.x, t = threadIdx.x;
  sWs[t] = Ws[t];
  sWs[t + 256] = Ws[t + 256];
  __syncthreads();

  const float* xb = x + (size_t)b * 32768;
  const int oy = t >> 4, ox = t & 15;
  float acc = bs[0];
  for (int c = 0; c < 32; ++c) {
    const float* xc = xb + c * 1024;
    const float* wc = sWs + c * 16;
#pragma unroll
    for (int ky = 0; ky < 4; ++ky) {
      const int iy = 2 * oy - 1 + ky;
      if ((unsigned)iy >= 32u) continue;
#pragma unroll
      for (int kx = 0; kx < 4; ++kx) {
        const int ix = 2 * ox - 1 + kx;
        if ((unsigned)ix >= 32u) continue;
        float v = xc[iy * 32 + ix];
        v = v > 0.f ? v : 0.f;
        acc += v * wc[ky * 4 + kx];
      }
    }
  }
  sS[t] = acc > 0.f ? acc : 0.f;
  __syncthreads();

  if (t < 12) {
    float a = bl[t];
    for (int k = 0; k < 256; ++k) a += sS[k] * Wl[k * 12 + t];
    sCtrl[t] = a;
  }
  __syncthreads();

  if (t == 0) {
    float yl[4], g[4], zm[4], zlv[4], e[4], z[4];
#pragma unroll
    for (int j = 0; j < 4; ++j) {
      yl[j] = sCtrl[j];
      zm[j] = sCtrl[4 + j];
      zlv[j] = sCtrl[8 + j];
      g[j] = yl[j] + gum[b * 4 + j];
    }
    float m = g[0];
    int idx = 0;
#pragma unroll
    for (int j = 1; j < 4; ++j)
      if (g[j] > m) { m = g[j]; idx = j; }   // first-max, like jnp.argmax
    float se = 0.f;
#pragma unroll
    for (int j = 0; j < 4; ++j) { e[j] = __expf(g[j] - m); se += e[j]; }
#pragma unroll
    for (int j = 0; j < 4; ++j)
      z[j] = eps[b * 4 + j] * __expf(zlv[j] * 0.5f) + zm[j];
#pragma unroll
    for (int j = 0; j < 4; ++j) {
      const float soft = e[j] / se;
      const float hard = (j == idx) ? 1.f : 0.f;
      out[O_Y + b * 4 + j] = yl[j];
      out[O_HARD + b * 4 + j] = (hard - soft) + soft;  // straight-through
    }
    out[O_IDX + b] = (float)idx;
    out[O_ZM + b] = zm[idx];
    out[O_ZLV + b] = zlv[idx];
    out[O_Z + b] = z[idx];
    ws_idx[b] = idx;
    ws_scale[b] = z[idx];
  }
}

// ---------------------------------------------------------------------------
// Kernel 2: switched expert convs via WMMA implicit GEMM + TDM staging.
// One block (256 thr = 8 waves) per batch element; selected expert only.
// ---------------------------------------------------------------------------
__global__ __launch_bounds__(256) void vae_sconv_kernel(
    const float* __restrict__ x, const float* __restrict__ W1,
    const float* __restrict__ b1, const float* __restrict__ W2,
    const float* __restrict__ b2, const int* __restrict__ ws_idx,
    const float* __restrict__ ws_scale, float* __restrict__ out) {
  extern __shared__ __align__(16) char smem[];
  float* xraw = (float*)smem;                          // 32768 f32 @ LDS 0
  _Float16* xpad = (_Float16*)(smem + LDS_RAW_BYTES);  // 34*34*32
  _Float16* hpad = xpad + 34 * 34 * 32;                // 34*34*8
  _Float16* w1t = hpad + 34 * 34 * 8;   // [f(9)][m(16)][c(32)], m>=8 zero
  _Float16* w2t = w1t + 9 * 16 * 32;    // [m(32)][k(96)], k=f*8+sm, k>=72 zero
  float* b1s = (float*)(w2t + 32 * 96);  // 8
  float* b2s = b1s + 8;                  // 32

  const int b = blockIdx.x;
  const int t = threadIdx.x;
  const int lane = t & 31;
  const int wave = t >> 5;
  const int bi = ws_idx[b];
  const float scale = ws_scale[b];
  const float* xb = x + (size_t)b * 32768;

  // ---- kick off TDM DMA of raw x[b] (128KB f32) into LDS, wave 0 only ----
  // D# per CDNA5 ISA 8.3/8.4: count=1, type=2(image), lds_addr=0,
  // data_size=4B, 1-D tile: tensor_dim0=tile_dim0=32768, dim1=1.
  if (wave == 0) {
    const unsigned long long ga = (unsigned long long)(uintptr_t)xb;
    v4u g0;
    g0[0] = 1u;                                       // count=1 (valid user D#)
    g0[1] = 0u;                                       // lds_addr = 0 (xraw)
    g0[2] = (unsigned)(ga & 0xFFFFFFFFu);             // global_addr[31:0]
    g0[3] = (unsigned)((ga >> 32) & 0x01FFFFFFu) | 0x80000000u;  // type=2
    v8i g1;
    g1[0] = 0x00020000;        // workgroup_mask=0, data_size=2 (4 bytes)
    g1[1] = (int)0x80000000;   // tensor_dim0[15:0]=0x8000 in bits[31:16]
    g1[2] = 0x00010000;        // tensor_dim0[31:16]=0, tensor_dim1[15:0]=1
    g1[3] = (int)0x80000000;   // tensor_dim1 hi=0, tile_dim0=0x8000
    g1[4] = 1;                 // tile_dim1=1, tile_dim2=0
    g1[5] = 32768;             // tensor_dim0_stride[31:0]
    g1[6] = 0;                 // stride hi, tensor_dim1_stride lo
    g1[7] = 0;
    v4i g2 = {0, 0, 0, 0};     // 2-D tensor: upper dims unused
    v4i g3 = {0, 0, 0, 0};
    v8i g4 = {0, 0, 0, 0, 0, 0, 0, 0};  // extra operand of 6-arg builtin
    __builtin_amdgcn_tensor_load_to_lds(g0, g1, g2, g3, g4, 0);
  }

  // ---- overlap with the DMA: zero halo'd LDS regions, stage weights ----
  {
    v8h zz = {};
    v8h* px8 = (v8h*)xpad;
    for (int i = t; i < (34 * 34 * 32) / 8; i += 256) px8[i] = zz;
    v8h* ph8 = (v8h*)hpad;
    for (int i = t; i < (34 * 34 * 8) / 8; i += 256) ph8[i] = zz;
  }
  {
    const float* w1 = W1 + (size_t)bi * (8 * 32 * 9);  // [o][c][f]
    for (int i = t; i < 9 * 16 * 32; i += 256) {
      const int cc = i & 31, fm = i >> 5, m = fm & 15, f = fm >> 4;
      w1t[i] = (m < 8) ? (_Float16)w1[(m * 32 + cc) * 9 + f] : (_Float16)0.f;
    }
    const float* w2 = W2 + (size_t)bi * (32 * 8 * 9);  // [o][sm][f]
    for (int i = t; i < 32 * 96; i += 256) {
      const int k = i % 96, m = i / 96, f = k >> 3, sm = k & 7;
      w2t[i] = (f < 9) ? (_Float16)w2[(m * 8 + sm) * 9 + f] : (_Float16)0.f;
    }
    if (t < 8) b1s[t] = b1[bi * 8 + t];
    if (t < 32) b2s[t] = b2[bi * 32 + t];
  }
  if (wave == 0) __builtin_amdgcn_s_wait_tensorcnt(0);
  __syncthreads();

  // ---- LDS->LDS convert: relu(x) f32 -> f16 halo layout, 4 px at a time ----
  for (int i = t * 4; i < 32768; i += 1024) {
    const int c = i >> 10, p = i & 1023;
    const int py = p >> 5, px = p & 31;
    const float4 v4 = *(const float4*)(xraw + i);
    _Float16* dst = &xpad[((py + 1) * 34 + (px + 1)) * 32 + c];
    dst[0]  = (_Float16)fmaxf(v4.x, 0.f);
    dst[32] = (_Float16)fmaxf(v4.y, 0.f);
    dst[64] = (_Float16)fmaxf(v4.z, 0.f);
    dst[96] = (_Float16)fmaxf(v4.w, 0.f);
  }
  __syncthreads();

  const int n = lane & 15;

  // ---- preload the 9 conv1 A fragments (16x32 f16 each) ----
  // A layout: lanes 0-15: K{0..15}; lanes 16-31: K{16..31}
  v16h a1[9];
  {
    const int cb = (lane < 16) ? 0 : 16;
    const int m = lane & 15;
#pragma unroll
    for (int f = 0; f < 9; ++f) {
      const _Float16* wr = &w1t[(f * 16 + m) * 32];
      v8h lo = *(const v8h*)(wr + cb);
      v8h hi = *(const v8h*)(wr + cb + 8);
      a1[f] = __builtin_shufflevector(lo, hi, 0, 1, 2, 3, 4, 5, 6, 7, 8, 9, 10,
                                      11, 12, 13, 14, 15);
    }
  }

  // ---- conv1: batch-issue all 18 ds_load_b128, fence the schedule, then ----
  // ---- run the 9-WMMA chain so LDS latency overlaps the matrix pipe.    ----
  // B layout: lanes 0-15 hold K=0..15 (channels), lanes 16-31 K=16..31
  for (int tile = wave * 8; tile < wave * 8 + 8; ++tile) {
    const int p = tile * 16 + n;
    const int py = p >> 5, px = p & 31;
    const int cb = (lane < 16) ? 0 : 16;
    v16h bf[9];
#pragma unroll
    for (int f = 0; f < 9; ++f) {
      const int dy = f / 3, dx = f % 3;
      const _Float16* src = &xpad[((py + dy) * 34 + (px + dx)) * 32 + cb];
      v8h lo = *(const v8h*)src;
      v8h hi = *(const v8h*)(src + 8);
      bf[f] = __builtin_shufflevector(lo, hi, 0, 1, 2, 3, 4, 5, 6, 7, 8, 9, 10,
                                      11, 12, 13, 14, 15);
    }
    __builtin_amdgcn_sched_barrier(0);  // keep all B loads ahead of the WMMAs
    v8f acc = {};
#pragma unroll
    for (int f = 0; f < 9; ++f)
      acc = __builtin_amdgcn_wmma_f32_16x16x32_f16(false, a1[f], false, bf[f],
                                                   (short)0, acc, false, false);
    // D rows: lanes 0-15 hold M=r (=sm r), N=lane -> bias+relu, store f16
    if (lane < 16) {
      v8h hv;
#pragma unroll
      for (int r = 0; r < 8; ++r) {
        float v = acc[r] + b1s[r];
        hv[r] = (_Float16)(v > 0.f ? v : 0.f);
      }
      *(v8h*)&hpad[((py + 1) * 34 + (px + 1)) * 8] = hv;
    }
  }

  // prefetch residual x for the conv2 writeback re-read (global_prefetch_b8)
  for (int i = t * 32; i < 32768; i += 256 * 32)
    __builtin_prefetch(xb + i, 0, 1);

  __syncthreads();

  // ---- conv2: preload all 6 tile-invariant A fragments, then stream tiles ---
  v16h a2[6];
#pragma unroll
  for (int mh = 0; mh < 2; ++mh)
#pragma unroll
    for (int ks = 0; ks < 3; ++ks) {
      const int m = mh * 16 + (lane & 15);
      const _Float16* wr = &w2t[m * 96 + ks * 32 + ((lane < 16) ? 0 : 16)];
      v8h lo = *(const v8h*)wr;
      v8h hi = *(const v8h*)(wr + 8);
      a2[mh * 3 + ks] = __builtin_shufflevector(lo, hi, 0, 1, 2, 3, 4, 5, 6, 7,
                                                8, 9, 10, 11, 12, 13, 14, 15);
    }

  for (int tile = wave * 8; tile < wave * 8 + 8; ++tile) {
    const int p = tile * 16 + n;
    const int py = p >> 5, px = p & 31;

    // B fragments from hpad: lanes 0-15: K=ks*32+0..15 -> offsets {4ks,4ks+1}
    //                        lanes16-31: K=ks*32+16..31 -> offsets {4ks+2,4ks+3}
    v16h bf[3];
#pragma unroll
    for (int ks = 0; ks < 3; ++ks) {
      int fa = ks * 4 + ((lane < 16) ? 0 : 2);
      int fb = fa + 1;
      if (fa > 8) fa = 8;  // padded K-range: A is zero there, value irrelevant
      if (fb > 8) fb = 8;
      const int dya = fa / 3, dxa = fa - dya * 3;
      const int dyb = fb / 3, dxb = fb - dyb * 3;
      v8h lo = *(const v8h*)&hpad[((py + dya) * 34 + (px + dxa)) * 8];
      v8h hi = *(const v8h*)&hpad[((py + dyb) * 34 + (px + dxb)) * 8];
      bf[ks] = __builtin_shufflevector(lo, hi, 0, 1, 2, 3, 4, 5, 6, 7, 8, 9, 10,
                                       11, 12, 13, 14, 15);
    }
    __builtin_amdgcn_sched_barrier(0);  // keep the 6 B loads ahead of WMMAs

#pragma unroll
    for (int mh = 0; mh < 2; ++mh) {
      v8f acc = {};
#pragma unroll
      for (int ks = 0; ks < 3; ++ks)
        acc = __builtin_amdgcn_wmma_f32_16x16x32_f16(
            false, a2[mh * 3 + ks], false, bf[ks], (short)0, acc, false, false);
      // writeback: out = x + scale*(conv2 + b2)
      const int mbase = mh * 16 + ((lane < 16) ? 0 : 8);
      const size_t obase = (size_t)b * 32768 + (size_t)p;
#pragma unroll
      for (int r = 0; r < 8; ++r) {
        const int m = mbase + r;
        const size_t o = obase + (size_t)m * 1024;
        out[o] = xb[m * 1024 + p] + scale * (acc[r] + b2s[m]);
      }
    }
  }
}

extern "C" void kernel_launch(void* const* d_in, const int* in_sizes, int n_in,
                              void* d_out, int out_size, void* d_ws,
                              size_t ws_size, hipStream_t stream) {
  const float* x   = (const float*)d_in[0];
  const float* gum = (const float*)d_in[1];
  const float* eps = (const float*)d_in[2];
  const float* W1  = (const float*)d_in[3];
  const float* b1  = (const float*)d_in[4];
  const float* W2  = (const float*)d_in[5];
  const float* b2  = (const float*)d_in[6];
  const float* Ws  = (const float*)d_in[7];
  const float* bs  = (const float*)d_in[8];
  const float* Wl  = (const float*)d_in[9];
  const float* bl  = (const float*)d_in[10];
  float* out = (float*)d_out;
  int* ws_idx = (int*)d_ws;
  float* ws_scale = (float*)((char*)d_ws + BATCH * sizeof(int));

  hipLaunchKernelGGL(vae_ctrl_kernel, dim3(BATCH), dim3(256), 0, stream, x, gum,
                     eps, Ws, bs, Wl, bl, out, ws_idx, ws_scale);

  hipLaunchKernelGGL(vae_sconv_kernel, dim3(BATCH), dim3(256),
                     (size_t)LDS_TOTAL_BYTES, stream, x, W1, b1, W2, b2, ws_idx,
                     ws_scale, out);
}
